// GraphNeuralSolver_24507083391633
// MI455X (gfx1250) — matrix operand order repaired and checked by reading, not compiled, not run on hardware
//
#include <hip/hip_runtime.h>

// ---------------------------------------------------------------------------
// Problem constants (from the reference)
// ---------------------------------------------------------------------------
#define KITER 3
#define NB    64
#define NPER  2000
#define EPER  4000
#define GPER  200
#define NN    (NB * NPER)   // 128000 buses
#define NE    (NB * EPER)   // 256000 edges
#define NG    (NB * GPER)   // 12800 generators
#define HD    128
#define HROW  129           // padded LDS hidden row stride (bank-conflict free)
#define LN_EPS 1e-5f

// Padded K dims (multiples of 32 for WMMA 16x16x32 bf16)
#define KP_PHI  160   // 133 -> 160
#define KP_NODE 288   // 260 -> 288

// Dynamic LDS budgets
#define PHI_LDS_BYTES  (KP_PHI * HD * 2 + HD * HD * 2 + 4 * 16 * HROW * 4)          // 106752
#define NODE_LDS_BYTES (KP_NODE * HD * 2 + HD * HD * 2 + 4 * 16 * HROW * 4 + 512)   // 140032

typedef __attribute__((ext_vector_type(16))) __bf16 bf16x16;
typedef __attribute__((ext_vector_type(8)))  float  f32x8;

union Frag { unsigned int u[8]; bf16x16 v; };

__device__ inline unsigned int packbf2(float a, float b) {
  unsigned int ba = __builtin_bit_cast(unsigned int, a);
  unsigned int bb = __builtin_bit_cast(unsigned int, b);
  ba = (ba + 0x7FFFu + ((ba >> 16) & 1u)) >> 16;      // RNE bf16
  bb = (bb + 0x7FFFu + ((bb >> 16) & 1u)) >> 16;
  return (ba & 0xFFFFu) | (bb << 16);
}
__device__ inline unsigned short bf16of(float a) {
  unsigned int ba = __builtin_bit_cast(unsigned int, a);
  return (unsigned short)((ba + 0x7FFFu + ((ba >> 16) & 1u)) >> 16);
}
__device__ inline void atomAdd(float* p, float v) {
  __hip_atomic_fetch_add(p, v, __ATOMIC_RELAXED, __HIP_MEMORY_SCOPE_AGENT);
}

// ---------------------------------------------------------------------------
// Cooperative global->LDS weight staging with 16B-granularity XOR swizzle.
// CPR = 16B chunks per row (row = CPR*16 bytes); power of two.
// ---------------------------------------------------------------------------
template <int CPR>
__device__ inline void stage_w(void* dst, const void* src, int bytes, int tid) {
  uint4* d = (uint4*)dst;
  const uint4* s = (const uint4*)src;
  int n = bytes >> 4;
  for (int i = tid; i < n; i += 128) {
    int row = i / CPR;
    int ch = i & (CPR - 1);
    __builtin_prefetch(s + i + 128, 0, 1);  // global_prefetch_b8 on weight stream
    d[row * CPR + (ch ^ (row & (CPR - 1)))] = s[i];
  }
}

// ---------------------------------------------------------------------------
// GEMM1 (x @ W1 + b1) -> LayerNorm(g,beta) -> leaky_relu(0.01), result in LDS.
// One wave handles a 16-row tile, full 128 hidden columns.
// A-fragment per CDNA5 ISA (16-bit A 16x32): lane = M + 16*Kgrp; VGPR j holds
// K pair {16*(j>>2) + 8*kgrp + 2*(j&3)}. xpair(k) returns a pre-packed bf16
// pair (elements k, k+1) as one dword -> no packing VALU in the hot loop.
// B (W1) comes from LDS (swizzled rows of 256B), gathered per element: column
// n = lane&15, K pairs in the same order as A.
// ---------------------------------------------------------------------------
template <typename XP>
__device__ inline void gemm1_ln_leaky(const XP& xpair,
                                      const unsigned short* Wl,  // LDS, swizzled
                                      const float* __restrict__ b1,
                                      const float* __restrict__ gam,
                                      const float* __restrict__ bet,
                                      float* Hl, int Kp, int lane) {
  f32x8 acc[8];
#pragma unroll
  for (int t = 0; t < 8; ++t)
#pragma unroll
    for (int j = 0; j < 8; ++j) acc[t][j] = 0.f;

  const int kgrp = (lane >> 4) & 1;
  for (int kc = 0; kc < Kp; kc += 32) {
    Frag a;
#pragma unroll
    for (int j = 0; j < 8; ++j) {
      int kk = kc + ((j >> 2) << 4) + (kgrp << 3) + ((j & 3) << 1);
      a.u[j] = xpair(kk);
    }
    const int n16 = lane & 15;
#pragma unroll
    for (int t = 0; t < 8; ++t) {
      Frag b;
#pragma unroll
      for (int j = 0; j < 8; ++j) {
        int k0 = kc + ((j >> 2) << 4) + (kgrp << 3) + ((j & 3) << 1);
        int n = t * 16 + n16;
        int ci0 = (((n >> 3) ^ (k0 & 15)) << 3) | (n & 7);
        int ci1 = (((n >> 3) ^ ((k0 + 1) & 15)) << 3) | (n & 7);
        b.u[j] = (unsigned int)Wl[k0 * HD + ci0] |
                 ((unsigned int)Wl[(k0 + 1) * HD + ci1] << 16);
      }
      acc[t] = __builtin_amdgcn_wmma_f32_16x16x32_bf16(
          false, a.v, false, b.v, (short)0, acc[t], false, false);
    }
  }
  // C layout: VGPR j -> row j + 8*kgrp, lane&15 -> col within 16-tile
  const int rofs = kgrp * 8;
  const int c0 = lane & 15;
#pragma unroll
  for (int t = 0; t < 8; ++t)
#pragma unroll
    for (int j = 0; j < 8; ++j) {
      int col = t * 16 + c0;
      Hl[(j + rofs) * HROW + col] = acc[t][j] + b1[col];
    }
  __syncthreads();
  if (lane < 16) {
    float* row = Hl + lane * HROW;
    float mu = 0.f;
    for (int i = 0; i < HD; ++i) mu += row[i];
    mu *= (1.f / HD);
    float var = 0.f;
    for (int i = 0; i < HD; ++i) { float d = row[i] - mu; var += d * d; }
    var *= (1.f / HD);
    float inv = rsqrtf(var + LN_EPS);
    for (int i = 0; i < HD; ++i) {
      float h = (row[i] - mu) * inv * gam[i] + bet[i];
      row[i] = (h > 0.f) ? h : 0.01f * h;
    }
  }
  __syncthreads();
}

// GEMM2: hidden(16x128 f32, LDS) @ W2(128 x NT*16 bf16, LDS swizzled rows)
template <int NT>
__device__ inline void gemm2(const float* Hl, const unsigned short* W2l,
                             f32x8* out, int lane) {
  const int kgrp = (lane >> 4) & 1;
  const int m = lane & 15;
#pragma unroll
  for (int t = 0; t < NT; ++t)
#pragma unroll
    for (int j = 0; j < 8; ++j) out[t][j] = 0.f;
  for (int kc = 0; kc < HD; kc += 32) {
    Frag a;
#pragma unroll
    for (int j = 0; j < 8; ++j) {
      int kk = kc + ((j >> 2) << 4) + (kgrp << 3) + ((j & 3) << 1);
      a.u[j] = packbf2(Hl[m * HROW + kk], Hl[m * HROW + kk + 1]);
    }
#pragma unroll
    for (int t = 0; t < NT; ++t) {
      Frag b;
#pragma unroll
      for (int j = 0; j < 8; ++j) {
        int k0 = kc + ((j >> 2) << 4) + (kgrp << 3) + ((j & 3) << 1);
        int n = t * 16 + m;
        int ci0 = (((n >> 3) ^ (k0 & (2 * NT - 1))) << 3) | (n & 7);
        int ci1 = (((n >> 3) ^ ((k0 + 1) & (2 * NT - 1))) << 3) | (n & 7);
        b.u[j] = (unsigned int)W2l[k0 * (NT * 16) + ci0] |
                 ((unsigned int)W2l[(k0 + 1) * (NT * 16) + ci1] << 16);
      }
      out[t] = __builtin_amdgcn_wmma_f32_16x16x32_bf16(
          false, a.v, false, b.v, (short)0, out[t], false, false);
    }
  }
}

// ---------------------------------------------------------------------------
// Utility kernels
// ---------------------------------------------------------------------------
__global__ void zero_kernel(float* p, long n) {
  long i = (long)blockIdx.x * blockDim.x + threadIdx.x;
  long st = (long)gridDim.x * blockDim.x;
  for (; i < n; i += st) p[i] = 0.f;
}

__global__ void copy_state_kernel(const float* __restrict__ v,
                                  const float* __restrict__ th,
                                  float* vcur, float* thcur) {
  int i = blockIdx.x * blockDim.x + threadIdx.x;
  if (i < NN) { vcur[i] = v[i]; thcur[i] = th[i]; }
}

__global__ void cvt_pad_kernel(const float* __restrict__ s,
                               unsigned short* __restrict__ d,
                               int K, int Ncol, int Kp, int Np) {
  int i = blockIdx.x * blockDim.x + threadIdx.x;
  if (i >= Kp * Np) return;
  int k = i / Np, n = i % Np;
  d[i] = (k < K && n < Ncol) ? bf16of(s[k * Ncol + n]) : (unsigned short)0;
}

__global__ void line16_kernel(const float* __restrict__ ea,
                              unsigned short* __restrict__ line16) {
  int e = blockIdx.x * blockDim.x + threadIdx.x;
  if (e >= NE) return;
  const float* a6 = ea + (size_t)e * 6;
  unsigned int* o = (unsigned int*)(line16 + (size_t)e * 32);
  o[0] = packbf2(a6[0], a6[1]);   // r, x
  o[1] = packbf2(a6[3], a6[4]);   // b_ij, tau
  o[2] = packbf2(a6[5], 0.f);     // shift, pad
#pragma unroll
  for (int i = 3; i < 16; ++i) o[i] = 0u;
}

__global__ void agg2bf_kernel(const float* __restrict__ agg,
                              unsigned int* __restrict__ agg16) {
  long n = (long)NN * HD / 2;
  long i = (long)blockIdx.x * blockDim.x + threadIdx.x;
  long st = (long)gridDim.x * blockDim.x;
  for (; i < n; i += st) agg16[i] = packbf2(agg[2 * i], agg[2 * i + 1]);
}

__global__ void gensums_kernel(const float* __restrict__ gg,
                               const float* __restrict__ gl,
                               const int* __restrict__ gb,
                               const int* __restrict__ slack,
                               float* pgsum, float* pgns, float* pgsp,
                               float* pgmin, float* pgmax) {
  int i = blockIdx.x * blockDim.x + threadIdx.x;
  if (i >= NG) return;
  float ps = gg[2 * i];
  float sl = slack[i] ? 1.f : 0.f;
  int b = gb[i];
  atomAdd(&pgsum[b], ps);
  atomAdd(&pgns[b], ps * (1.f - sl));
  atomAdd(&pgsp[b], ps * sl);
  atomAdd(&pgmin[b], gl[2 * i] * sl);
  atomAdd(&pgmax[b], gl[2 * i + 1] * sl);
}

__global__ void flows_kernel(const float* __restrict__ vcur,
                             const float* __restrict__ thcur,
                             const float* __restrict__ ea,
                             const int* __restrict__ src,
                             const int* __restrict__ dst,
                             const int* __restrict__ bus_batch,
                             float* Pbus, float* Qbus, float* pj) {
  int e = blockIdx.x * blockDim.x + threadIdx.x;
  if (e >= NE) return;
  const float* a6 = ea + (size_t)e * 6;
  float r = a6[0], x = a6[1], bsh = a6[3], tau = a6[4], sh = a6[5];
  float y = 1.f / sqrtf(r * r + x * x);
  float del = -atan2f(x, r);
  int si = src[e], di = dst[e];
  float vi = vcur[si], vj = vcur[di], ti = thcur[si], tj = thcur[di];
  float a = vi * vj * y / tau;
  float sd = sinf(del), cd = cosf(del);
  float vit = vi / tau;
  float Ps = a * sinf(ti - tj - del - sh) + vit * vit * (y * sd);
  float Pd = a * sinf(tj - ti - del + sh) + vj * vj * (y * sd);
  float Qs = -a * cosf(ti - tj - del - sh) + vit * vit * (y * cd - 0.5f * bsh);
  float Qd = -a * cosf(tj - ti - del + sh) + vj * vj * (y * sd - 0.5f * bsh);
  atomAdd(&Pbus[si], Ps);
  atomAdd(&Pbus[di], Pd);
  atomAdd(&Qbus[si], Qs);
  atomAdd(&Qbus[di], Qd);
  atomAdd(&pj[bus_batch[si]], fabsf(Ps + Pd));
}

__global__ void busload_kernel(const float* __restrict__ vcur,
                               const float* __restrict__ pd,
                               const float* __restrict__ shunt,
                               const int* __restrict__ bus_batch, float* pglob) {
  int n = blockIdx.x * blockDim.x + threadIdx.x;
  if (n >= NN) return;
  float vv = vcur[n];
  atomAdd(&pglob[bus_batch[n]], pd[n] + vv * vv * shunt[2 * n]);
}

__global__ void lamb_kernel(const float* pglob_a, const float* pj,
                            const float* pgsum, const float* pgns,
                            const float* pgsp, const float* pgmin,
                            const float* pgmax, float* pgslack) {
  int b = threadIdx.x;
  if (b >= NB) return;
  float pglob = pglob_a[b] + pj[b];
  float sum = pgsum[b], ns = pgns[b], sp = pgsp[b], mn = pgmin[b], mx = pgmax[b];
  float lamb = (pglob < sum) ? (pglob - ns - mx) / (2.f * (sp - mn))
                             : (pglob - ns - 2.f * sp - mx) / (2.f * (mx - sp));
  lamb = fmaxf(lamb, 0.f);
  pgslack[b] = (lamb < 0.5f) ? mn + 2.f * (sp - mn) * lamb
                             : 2.f * sp - mx + 2.f * (mx - sp) * lamb;
}

// Mismatch kernel: also emits the packed bf16 scalar-feature quad [v,th,dp,dq]
__global__ void deltas_kernel(const float* __restrict__ vcur,
                              const float* __restrict__ thcur,
                              const float* __restrict__ pg,
                              const float* __restrict__ pd,
                              const float* __restrict__ qd,
                              const float* __restrict__ shunt,
                              const float* __restrict__ Pbus,
                              const float* __restrict__ Qbus,
                              const int* __restrict__ bus_type,
                              const int* __restrict__ bus_batch,
                              const float* __restrict__ pgslack,
                              unsigned short* __restrict__ feat4, float* dsum) {
  int n = blockIdx.x * blockDim.x + threadIdx.x;
  if (n >= NN) return;
  float vv = vcur[n], v2 = vv * vv;
  float gs = shunt[2 * n], bs = shunt[2 * n + 1];
  float pge = (bus_type[n] == 3) ? pgslack[bus_batch[n]] : pg[n];
  float dpp = pge - pd[n] - gs * v2 + Pbus[n];
  float qg = qd[n] - bs * v2 - Qbus[n];
  float dqq = qg - qd[n] + bs * v2 + Qbus[n];
  unsigned int* o = (unsigned int*)(feat4 + (size_t)n * 4);
  o[0] = packbf2(vv, thcur[n]);
  o[1] = packbf2(dpp, dqq);
  atomAdd(dsum, dpp * dpp + dqq * dqq);
}

__global__ void total_update_kernel(float* total, const float* dsum, float gw) {
  if (threadIdx.x == 0 && blockIdx.x == 0)
    total[0] += dsum[0] * (1.f / (float)NN) * gw;
}

// ---------------------------------------------------------------------------
// phi MLP over edges: msgs = phi([m[src], line]); agg[dst] += msgs (atomic).
// Weights staged in LDS once per block; A-operands are pre-packed bf16 pairs.
// ---------------------------------------------------------------------------
__global__ void __launch_bounds__(128)
phi_edge_kernel(const unsigned short* __restrict__ m_bf16,
                const unsigned short* __restrict__ line16,
                const int* __restrict__ src, const int* __restrict__ dst,
                const unsigned short* __restrict__ W1g,
                const float* __restrict__ b1, const float* __restrict__ g,
                const float* __restrict__ be,
                const unsigned short* __restrict__ W2g,
                const float* __restrict__ b2, float* __restrict__ agg) {
  extern __shared__ char smem[];
  unsigned short* W1l = (unsigned short*)smem;                       // 40960 B
  unsigned short* W2l = (unsigned short*)(smem + KP_PHI * HD * 2);   // 32768 B
  float* Hall = (float*)(smem + KP_PHI * HD * 2 + HD * HD * 2);
  int tid = threadIdx.x, wave = tid >> 5, lane = tid & 31;
  float* Hl = Hall + wave * 16 * HROW;

  stage_w<16>(W1l, W1g, KP_PHI * HD * 2, tid);
  stage_w<16>(W2l, W2g, HD * HD * 2, tid);
  __syncthreads();

  long e0 = ((long)blockIdx.x * 4 + wave) * 16;
  long e = e0 + (lane & 15);
  const unsigned short* mrow16 = m_bf16 + (size_t)src[e] * HD;
  const unsigned short* lrow16 = line16 + (size_t)e * 32;
  auto xpair = [&](int k) -> unsigned int {
    return (k < HD) ? *(const unsigned int*)(mrow16 + k)
                    : *(const unsigned int*)(lrow16 + (k - HD));
  };
  gemm1_ln_leaky(xpair, W1l, b1, g, be, Hl, KP_PHI, lane);
  f32x8 out[8];
  gemm2<8>(Hl, W2l, out, lane);
  int rofs = ((lane >> 4) & 1) * 8;
  int c0 = lane & 15;
#pragma unroll
  for (int t = 0; t < 8; ++t)
#pragma unroll
    for (int j = 0; j < 8; ++j) {
      int col = t * 16 + c0;
      int dn = dst[e0 + j + rofs];
      atomAdd(&agg[(size_t)dn * HD + col], out[t][j] + b2[col]);
    }
}

// ---------------------------------------------------------------------------
// Node update: feat = [v, th, dp, dq, m, agg]; theta += t(feat);
// v += v(feat) (non-PV); m += m(feat). All three MLPs see the same feat.
// ---------------------------------------------------------------------------
struct MlpPtrs {
  const unsigned short* W1;
  const float* b1;
  const float* g;
  const float* be;
  const unsigned short* W2;
  const float* b2;
};

__global__ void __launch_bounds__(128)
node_update_kernel(float* __restrict__ vcur, float* __restrict__ thcur,
                   const unsigned short* __restrict__ feat4,
                   float* __restrict__ m_state,
                   unsigned short* __restrict__ m_bf16,
                   const unsigned short* __restrict__ agg16,
                   const int* __restrict__ bus_type,
                   MlpPtrs T, MlpPtrs V, MlpPtrs M) {
  extern __shared__ char smem[];
  unsigned short* W1l = (unsigned short*)smem;                        // 73728 B
  unsigned short* W2l = (unsigned short*)(smem + KP_NODE * HD * 2);   // 32768 B
  float* Hall = (float*)(smem + KP_NODE * HD * 2 + HD * HD * 2);
  float* tvb = (float*)(smem + KP_NODE * HD * 2 + HD * HD * 2 + 4 * 16 * HROW * 4);
  int tid = threadIdx.x, wave = tid >> 5, lane = tid & 31;
  float* Hl = Hall + wave * 16 * HROW;
  float* tval = tvb + wave * 16;
  float* vval = tvb + 64 + wave * 16;

  long n0 = ((long)blockIdx.x * 4 + wave) * 16;
  long n = n0 + (lane & 15);
  float f0 = vcur[n], f1 = thcur[n];
  const unsigned short* f4 = feat4 + (size_t)n * 4;
  const unsigned short* mrow16 = m_bf16 + (size_t)n * HD;
  const unsigned short* arow16 = agg16 + (size_t)n * HD;
  auto xpair = [&](int k) -> unsigned int {
    if (k < 4) return *(const unsigned int*)(f4 + k);
    if (k < 4 + HD) return *(const unsigned int*)(mrow16 + (k - 4));
    if (k < 4 + 2 * HD) return *(const unsigned int*)(arow16 + (k - 4 - HD));
    return 0u;
  };
  int rofs = ((lane >> 4) & 1) * 8;
  int c0 = lane & 15;

  // ---- t-MLP (scalar output, col 0 of 16-wide padded W2)
  stage_w<16>(W1l, T.W1, KP_NODE * HD * 2, tid);
  stage_w<2>(W2l, T.W2, HD * 16 * 2, tid);
  __syncthreads();
  gemm1_ln_leaky(xpair, W1l, T.b1, T.g, T.be, Hl, KP_NODE, lane);
  {
    f32x8 o[1];
    gemm2<1>(Hl, W2l, o, lane);
    if (c0 == 0)
#pragma unroll
      for (int j = 0; j < 8; ++j) tval[j + rofs] = o[0][j] + T.b2[0];
  }
  __syncthreads();

  // ---- v-MLP
  stage_w<16>(W1l, V.W1, KP_NODE * HD * 2, tid);
  stage_w<2>(W2l, V.W2, HD * 16 * 2, tid);
  __syncthreads();
  gemm1_ln_leaky(xpair, W1l, V.b1, V.g, V.be, Hl, KP_NODE, lane);
  {
    f32x8 o[1];
    gemm2<1>(Hl, W2l, o, lane);
    if (c0 == 0)
#pragma unroll
      for (int j = 0; j < 8; ++j) vval[j + rofs] = o[0][j] + V.b2[0];
  }
  __syncthreads();

  // ---- m-MLP (128-wide output)
  stage_w<16>(W1l, M.W1, KP_NODE * HD * 2, tid);
  stage_w<16>(W2l, M.W2, HD * HD * 2, tid);
  __syncthreads();
  gemm1_ln_leaky(xpair, W1l, M.b1, M.g, M.be, Hl, KP_NODE, lane);
  f32x8 o8[8];
  gemm2<8>(Hl, W2l, o8, lane);
#pragma unroll
  for (int t = 0; t < 8; ++t)
#pragma unroll
    for (int j = 0; j < 8; ++j) {
      long rr = n0 + j + rofs;
      int col = t * 16 + c0;
      size_t idx = (size_t)rr * HD + col;
      float nv = m_state[idx] + o8[t][j] + M.b2[col];
      m_state[idx] = nv;
      m_bf16[idx] = bf16of(nv);  // keep bf16 mirror in sync for next round
    }
  __syncthreads();

  if (lane < 16) {
    long r = n0 + lane;
    thcur[r] = f1 + tval[lane];
    if (bus_type[r] != 2) vcur[r] = f0 + vval[lane];
  }
}

__global__ void emit_out_kernel(float* __restrict__ out,
                                const float* __restrict__ vcur,
                                const float* __restrict__ thcur,
                                const float* __restrict__ m_state,
                                const float* __restrict__ total) {
  long tot = 2L * NN + (long)NN * HD + 1;
  long i = (long)blockIdx.x * blockDim.x + threadIdx.x;
  long st = (long)gridDim.x * blockDim.x;
  for (; i < tot; i += st) {
    if (i < NN) out[i] = vcur[i];
    else if (i < 2 * NN) out[i] = thcur[i - NN];
    else if (i < 2L * NN + (long)NN * HD) out[i] = m_state[i - 2 * NN];
    else out[i] = total[0];
  }
}

// ---------------------------------------------------------------------------
// Host-side orchestration
// ---------------------------------------------------------------------------
extern "C" void kernel_launch(void* const* d_in, const int* in_sizes, int n_in,
                              void* d_out, int out_size, void* d_ws, size_t ws_size,
                              hipStream_t stream) {
  // Input flattening assumption (JAX pytree order, dict keys sorted):
  //  params.layers[k] -> {m,t,v} each {b1,b2,beta,g,w1,w2}: leaves k*18 + which*6 + {0..5}
  //  params.phi at 54..59, then v=60, theta=61, pg=62, pd=63, qd=64, shunt=65,
  //  edge_attr=66, gen_generation=67, gen_limits=68, src=69, dst=70,
  //  bus_batch=71, gen_batch=72, is_slack=73, bus_type=74.
  auto F = [&](int i) { return (const float*)d_in[i]; };
  auto I = [&](int i) { return (const int*)d_in[i]; };

  const float* in_v = F(60);
  const float* in_th = F(61);
  const float* in_pg = F(62);
  const float* in_pd = F(63);
  const float* in_qd = F(64);
  const float* in_shunt = F(65);
  const float* in_ea = F(66);
  const float* in_gg = F(67);
  const float* in_gl = F(68);
  const int* in_src = I(69);
  const int* in_dst = I(70);
  const int* in_bb = I(71);
  const int* in_gb = I(72);
  const int* in_sl = I(73);
  const int* in_bt = I(74);

  // Workspace carve-up
  size_t cur = 0;
  auto alloc = [&](size_t bytes) -> void* {
    size_t a = (cur + 255) & ~(size_t)255;
    cur = a + bytes;
    return (void*)((char*)d_ws + a);
  };
  unsigned short* phiW1 = (unsigned short*)alloc((size_t)KP_PHI * HD * 2);
  unsigned short* phiW2 = (unsigned short*)alloc((size_t)HD * HD * 2);
  unsigned short *tW1[KITER], *tW2[KITER], *vW1[KITER], *vW2[KITER],
      *mW1[KITER], *mW2[KITER];
  for (int k = 0; k < KITER; ++k) {
    mW1[k] = (unsigned short*)alloc((size_t)KP_NODE * HD * 2);
    mW2[k] = (unsigned short*)alloc((size_t)HD * HD * 2);
    tW1[k] = (unsigned short*)alloc((size_t)KP_NODE * HD * 2);
    tW2[k] = (unsigned short*)alloc((size_t)HD * 16 * 2);
    vW1[k] = (unsigned short*)alloc((size_t)KP_NODE * HD * 2);
    vW2[k] = (unsigned short*)alloc((size_t)HD * 16 * 2);
  }
  float* vcur = (float*)alloc(NN * 4);
  float* thcur = (float*)alloc(NN * 4);
  float* Pbus = (float*)alloc((size_t)2 * NN * 4);  // Pbus|Qbus contiguous
  float* Qbus = Pbus + NN;
  float* pj = (float*)alloc((2 * NB + 1) * 4);      // pj|pglob|dsum contiguous
  float* pglob = pj + NB;
  float* dsum = pj + 2 * NB;
  float* pgsum = (float*)alloc((5 * NB + 1) * 4);   // 5 sums | total contiguous
  float* pgns = pgsum + NB;
  float* pgsp = pgsum + 2 * NB;
  float* pgmin = pgsum + 3 * NB;
  float* pgmax = pgsum + 4 * NB;
  float* total = pgsum + 5 * NB;
  float* pgslack = (float*)alloc(NB * 4);
  float* m_state = (float*)alloc((size_t)NN * HD * 4);
  float* agg = (float*)alloc((size_t)NN * HD * 4);
  unsigned short* m_bf16 = (unsigned short*)alloc((size_t)NN * HD * 2);
  unsigned short* agg16 = (unsigned short*)alloc((size_t)NN * HD * 2);
  unsigned short* line16 = (unsigned short*)alloc((size_t)NE * 32 * 2);
  unsigned short* feat4 = (unsigned short*)alloc((size_t)NN * 4 * 2);
  (void)ws_size; (void)in_sizes; (void)n_in; (void)out_size;

  // Allow >64KB dynamic LDS for the WMMA kernels (host-side, capture-safe)
  (void)hipFuncSetAttribute(reinterpret_cast<const void*>(phi_edge_kernel),
                            hipFuncAttributeMaxDynamicSharedMemorySize,
                            PHI_LDS_BYTES);
  (void)hipFuncSetAttribute(reinterpret_cast<const void*>(node_update_kernel),
                            hipFuncAttributeMaxDynamicSharedMemorySize,
                            NODE_LDS_BYTES);

  // --- one-time init ---
  zero_kernel<<<4096, 256, 0, stream>>>(m_state, (long)NN * HD);
  zero_kernel<<<4096, 256, 0, stream>>>((float*)m_bf16, (long)NN * HD / 2);
  zero_kernel<<<2, 256, 0, stream>>>(pgsum, 5 * NB + 1);
  copy_state_kernel<<<(NN + 255) / 256, 256, 0, stream>>>(in_v, in_th, vcur, thcur);
  gensums_kernel<<<(NG + 255) / 256, 256, 0, stream>>>(in_gg, in_gl, in_gb, in_sl,
                                                       pgsum, pgns, pgsp, pgmin, pgmax);
  line16_kernel<<<(NE + 255) / 256, 256, 0, stream>>>(in_ea, line16);
  auto cvt = [&](const float* s, unsigned short* dst_, int K, int Ncol, int Kp, int Np) {
    int tot = Kp * Np;
    cvt_pad_kernel<<<(tot + 255) / 256, 256, 0, stream>>>(s, dst_, K, Ncol, Kp, Np);
  };
  cvt(F(58), phiW1, 133, HD, KP_PHI, HD);
  cvt(F(59), phiW2, HD, HD, HD, HD);
  for (int k = 0; k < KITER; ++k) {
    int bm = k * 18, bt = k * 18 + 6, bv = k * 18 + 12;
    cvt(F(bm + 4), mW1[k], 260, HD, KP_NODE, HD);
    cvt(F(bm + 5), mW2[k], HD, HD, HD, HD);
    cvt(F(bt + 4), tW1[k], 260, HD, KP_NODE, HD);
    cvt(F(bt + 5), tW2[k], HD, 1, HD, 16);
    cvt(F(bv + 4), vW1[k], 260, HD, KP_NODE, HD);
    cvt(F(bv + 5), vW2[k], HD, 1, HD, 16);
  }

  const float gw[KITER] = {0.729f, 0.81f, 0.9f};  // GAMMA^(K-k)
  for (int k = 0; k < KITER; ++k) {
    zero_kernel<<<2048, 256, 0, stream>>>(Pbus, (long)2 * NN);
    zero_kernel<<<2, 256, 0, stream>>>(pj, 2 * NB + 1);
    flows_kernel<<<(NE + 255) / 256, 256, 0, stream>>>(vcur, thcur, in_ea, in_src,
                                                       in_dst, in_bb, Pbus, Qbus, pj);
    busload_kernel<<<(NN + 255) / 256, 256, 0, stream>>>(vcur, in_pd, in_shunt,
                                                         in_bb, pglob);
    lamb_kernel<<<1, NB, 0, stream>>>(pglob, pj, pgsum, pgns, pgsp, pgmin, pgmax,
                                      pgslack);
    deltas_kernel<<<(NN + 255) / 256, 256, 0, stream>>>(vcur, thcur, in_pg, in_pd,
                                                        in_qd, in_shunt, Pbus, Qbus,
                                                        in_bt, in_bb, pgslack,
                                                        feat4, dsum);
    total_update_kernel<<<1, 1, 0, stream>>>(total, dsum, gw[k]);
    zero_kernel<<<4096, 256, 0, stream>>>(agg, (long)NN * HD);

    phi_edge_kernel<<<NE / 64, 128, PHI_LDS_BYTES, stream>>>(
        m_bf16, line16, in_src, in_dst, phiW1, F(54), F(57), F(56), phiW2, F(55),
        agg);
    agg2bf_kernel<<<4096, 256, 0, stream>>>(agg, (unsigned int*)agg16);

    int bm = k * 18, bt = k * 18 + 6, bv = k * 18 + 12;
    MlpPtrs T{tW1[k], F(bt + 0), F(bt + 3), F(bt + 2), tW2[k], F(bt + 1)};
    MlpPtrs V{vW1[k], F(bv + 0), F(bv + 3), F(bv + 2), vW2[k], F(bv + 1)};
    MlpPtrs M{mW1[k], F(bm + 0), F(bm + 3), F(bm + 2), mW2[k], F(bm + 1)};
    node_update_kernel<<<NN / 64, 128, NODE_LDS_BYTES, stream>>>(
        vcur, thcur, feat4, m_state, m_bf16, agg16, in_bt, T, V, M);
  }
  emit_out_kernel<<<4096, 256, 0, stream>>>((float*)d_out, vcur, thcur, m_state,
                                            total);
}
